// TransformerBlock_37787122270324
// MI455X (gfx1250) — compile-verified, matrix operand order checked by reading, not run on hardware
//
#include <hip/hip_runtime.h>
#include <cstdint>
#include <cstddef>

// ---------------------------------------------------------------------------
// MI455X (gfx1250) MoE transformer block.
// All large GEMMs: v_wmma_f32_16x16x32_bf16, fp32 accumulate.
// Every GEMM operand is K-contiguous (B pre-transposed to [N,K]) so each
// WMMA fragment is exactly two 16-byte loads per lane (ISA §7.12.2 layout).
// LDS staging uses global_load_async_to_lds_b128 + s_wait_asynccnt.
// ---------------------------------------------------------------------------

typedef __bf16 bf16_t;
typedef __bf16   v16bf __attribute__((ext_vector_type(16)));
typedef float    v8f   __attribute__((ext_vector_type(8)));
typedef uint32_t u32x4 __attribute__((ext_vector_type(4)));
typedef int      v4i_t __attribute__((ext_vector_type(4)));

#define BDIM 256                 // 8 waves / block (wave32)

#if defined(__gfx1250__) && __has_builtin(__builtin_amdgcn_global_load_async_to_lds_b128) && __has_builtin(__builtin_amdgcn_s_wait_asynccnt)
#define USE_ASYNC_LDS 1
#else
#define USE_ASYNC_LDS 0
#endif

// ---- helpers ---------------------------------------------------------------

__device__ __forceinline__ bf16_t f2bf(float f) {
  uint32_t u = __builtin_bit_cast(uint32_t, f);
  u = u + 0x7FFFu + ((u >> 16) & 1u);            // round-to-nearest-even
  unsigned short s = (unsigned short)(u >> 16);
  return __builtin_bit_cast(bf16_t, s);
}

__device__ __forceinline__ u32x4 ld128(const bf16_t* p) {
  return *reinterpret_cast<const u32x4*>(p);
}

// Two 16-byte chunks -> one 16x32-step bf16 WMMA fragment (32 bytes).
__device__ __forceinline__ v16bf make_frag(u32x4 a, u32x4 b) {
  union { struct { u32x4 x, y; } s; v16bf v; } u;
  u.s.x = a; u.s.y = b;
  return u.v;
}

// Copy 16 bytes global -> LDS (async on CDNA5 when available).
__device__ __forceinline__ void cp16_g2l(const bf16_t* g, bf16_t* l) {
#if USE_ASYNC_LDS
  __builtin_amdgcn_global_load_async_to_lds_b128(
      (__attribute__((address_space(1))) v4i_t*)g,
      (__attribute__((address_space(3))) v4i_t*)l, 0, 0);
#else
  *reinterpret_cast<u32x4*>(l) = ld128(g);
#endif
}

__device__ __forceinline__ void async_fence() {
#if USE_ASYNC_LDS
  __builtin_amdgcn_s_wait_asynccnt(0);
#endif
  __syncthreads();
}

// ---------------------------------------------------------------------------
// Block-cooperative WMMA GEMM with LDS staging.
//   C[M,N] = A[M,K] (bf16, row-major, lda) * Bt[N,K]^T (bf16, row-major, ldb)
// Block tile 128x64, wave tile 32x32 (2x2 WMMA tiles), K step 32.
// Requires M % 128 == 0, N % 64 == 0, K % 32 == 0 (all call sites satisfy).
// EPI: 0 = store f32, 1 = accumulate f32 (C+=), 2 = ReLU->bf16, 3 = bf16
// Batched via blockIdx.z = bi*batchH + hb with independent element strides.
// ---------------------------------------------------------------------------
template <int EPI>
__global__ __launch_bounds__(BDIM)
void gemm_lds(const bf16_t* __restrict__ A, const bf16_t* __restrict__ Bt,
              void* __restrict__ Cv,
              int M, int N, int K, int lda, int ldb, int ldc,
              int batchH, long sAb, long sAh, long sBb, long sBh,
              long sCb, long sCh)
{
  __shared__ __align__(16) bf16_t As[128 * 32];   // 8 KB
  __shared__ __align__(16) bf16_t Bs[64 * 32];    // 4 KB

  const int z  = blockIdx.z;
  const int bi = z / batchH;
  const int hb = z % batchH;
  A  += (size_t)(bi * sAb + hb * sAh);
  Bt += (size_t)(bi * sBb + hb * sBh);

  const int tn_cnt = N >> 6;
  const int mBase  = (blockIdx.x / tn_cnt) << 7;  // *128
  const int nBase  = (blockIdx.x % tn_cnt) << 6;  // *64

  const int t    = threadIdx.x;
  const int w    = t >> 5;
  const int wr   = w >> 1;          // 0..3 : 32-row slab
  const int wc   = w & 1;           // 0..1 : 32-col slab
  const int lane = t & 31;
  const int l15  = lane & 15;
  const int half = lane >> 4;

  // staging chunk coordinates (16B chunks of the 32-wide K slab)
  const int sr = t >> 2;            // 0..63
  const int sc = (t & 3) * 8;       // 0,8,16,24 (bf16 elements)

  v8f acc[2][2];
  #pragma unroll
  for (int i = 0; i < 2; ++i)
    #pragma unroll
    for (int j = 0; j < 2; ++j)
      #pragma unroll
      for (int v = 0; v < 8; ++v) acc[i][j][v] = 0.0f;

  for (int kb = 0; kb < K; kb += 32) {
    // ---- stage A (128x32) : 2 chunks per thread; B (64x32) : 1 chunk ----
    cp16_g2l(A  + (size_t)(mBase + sr)      * lda + kb + sc, &As[sr * 32 + sc]);
    cp16_g2l(A  + (size_t)(mBase + sr + 64) * lda + kb + sc, &As[(sr + 64) * 32 + sc]);
    cp16_g2l(Bt + (size_t)(nBase + sr)      * ldb + kb + sc, &Bs[sr * 32 + sc]);
    async_fence();

    // ---- fragments from LDS (2x ds_load_b128 each) ----
    v16bf afr[2], bfr[2];
    #pragma unroll
    for (int i = 0; i < 2; ++i) {
      const bf16_t* ar = &As[(wr * 32 + i * 16 + l15) * 32];
      afr[i] = make_frag(ld128(ar + half * 8), ld128(ar + 16 + half * 8));
    }
    #pragma unroll
    for (int j = 0; j < 2; ++j) {
      const bf16_t* br = &Bs[(wc * 32 + j * 16 + l15) * 32];
      bfr[j] = make_frag(ld128(br + half * 8), ld128(br + 16 + half * 8));
    }
    #pragma unroll
    for (int i = 0; i < 2; ++i)
      #pragma unroll
      for (int j = 0; j < 2; ++j)
        acc[i][j] = __builtin_amdgcn_wmma_f32_16x16x32_bf16(
            false, afr[i], false, bfr[j], (short)0, acc[i][j], false, false);

    __syncthreads();
  }

  // ---- epilogue ----
  #pragma unroll
  for (int i = 0; i < 2; ++i) {
    #pragma unroll
    for (int j = 0; j < 2; ++j) {
      const int ncol = nBase + wc * 32 + j * 16 + l15;
      if (EPI == 0 || EPI == 1) {
        float* C = (float*)Cv + (size_t)(bi * sCb + hb * sCh);
        #pragma unroll
        for (int v = 0; v < 8; ++v) {
          const int m2 = mBase + wr * 32 + i * 16 + (half << 3) + v;
          const size_t idx = (size_t)m2 * ldc + ncol;
          if (EPI == 0) C[idx] = acc[i][j][v];
          else          C[idx] += acc[i][j][v];
        }
      } else {
        bf16_t* C = (bf16_t*)Cv + (size_t)(bi * sCb + hb * sCh);
        #pragma unroll
        for (int v = 0; v < 8; ++v) {
          const int m2 = mBase + wr * 32 + i * 16 + (half << 3) + v;
          float xv = acc[i][j][v];
          if (EPI == 2) xv = fmaxf(xv, 0.0f);
          C[(size_t)m2 * ldc + ncol] = f2bf(xv);
        }
      }
    }
  }
}

// ---------------------------------------------------------------------------
// Direct-from-global WMMA GEMM (for shapes not divisible by 64 in N).
// Wave tile 32x32 (2x2 WMMA). B is [N,K] (transposed weights).
// ---------------------------------------------------------------------------
template <int EPI>
__global__ __launch_bounds__(BDIM)
void gemm_direct(const bf16_t* __restrict__ A, const bf16_t* __restrict__ Bt,
                 void* __restrict__ Cv, int M, int N, int K,
                 int lda, int ldb, int ldc)
{
  const int tn_cnt = N >> 5;
  const int tile   = blockIdx.x * (BDIM / 32) + (threadIdx.x >> 5);
  if (tile >= (M >> 5) * tn_cnt) return;
  const int tm = (tile / tn_cnt) << 5;
  const int tn = (tile % tn_cnt) << 5;

  const int lane = threadIdx.x & 31;
  const int l15  = lane & 15;
  const int half = lane >> 4;

  v8f acc[2][2];
  #pragma unroll
  for (int i = 0; i < 2; ++i)
    #pragma unroll
    for (int j = 0; j < 2; ++j)
      #pragma unroll
      for (int v = 0; v < 8; ++v) acc[i][j][v] = 0.0f;

  const bf16_t* ar0 = A  + (size_t)(tm + l15)      * lda;
  const bf16_t* ar1 = A  + (size_t)(tm + 16 + l15) * lda;
  const bf16_t* br0 = Bt + (size_t)(tn + l15)      * ldb;
  const bf16_t* br1 = Bt + (size_t)(tn + 16 + l15) * ldb;

  for (int kb = 0; kb < K; kb += 32) {
    if (kb + 32 < K) __builtin_prefetch(ar0 + kb + 32, 0, 3);   // global_prefetch_b8
    v16bf afr[2], bfr[2];
    afr[0] = make_frag(ld128(ar0 + kb + half * 8), ld128(ar0 + kb + 16 + half * 8));
    afr[1] = make_frag(ld128(ar1 + kb + half * 8), ld128(ar1 + kb + 16 + half * 8));
    bfr[0] = make_frag(ld128(br0 + kb + half * 8), ld128(br0 + kb + 16 + half * 8));
    bfr[1] = make_frag(ld128(br1 + kb + half * 8), ld128(br1 + kb + 16 + half * 8));
    #pragma unroll
    for (int i = 0; i < 2; ++i)
      #pragma unroll
      for (int j = 0; j < 2; ++j)
        acc[i][j] = __builtin_amdgcn_wmma_f32_16x16x32_bf16(
            false, afr[i], false, bfr[j], (short)0, acc[i][j], false, false);
  }

  #pragma unroll
  for (int i = 0; i < 2; ++i)
    #pragma unroll
    for (int j = 0; j < 2; ++j) {
      const int ncol = tn + j * 16 + l15;
      #pragma unroll
      for (int v = 0; v < 8; ++v) {
        const int m2 = tm + i * 16 + (half << 3) + v;
        float* C = (float*)Cv;
        C[(size_t)m2 * ldc + ncol] = acc[i][j][v];   // EPI==0 only use
      }
    }
}

// ---------------------------------------------------------------------------
// Small utility kernels
// ---------------------------------------------------------------------------
__global__ void zero_k(float* p, long n) {
  long i = (long)blockIdx.x * blockDim.x + threadIdx.x;
  if (i < n) p[i] = 0.0f;
}

// s[K,N] (f32) -> d[N,K] (bf16), batched (per-matrix transpose).
__global__ void cvt_t_k(const float* __restrict__ s, bf16_t* __restrict__ d,
                        long K, long N, long nb)
{
  long i = (long)blockIdx.x * blockDim.x + threadIdx.x;
  const long per = K * N;
  if (i >= nb * per) return;
  const long b = i / per, r = i % per;
  const long n = r / K, k = r % K;
  d[i] = f2bf(s[b * per + k * N + n]);
}

// LayerNorm over D=768, one block per row; writes f32 + bf16 copies.
__global__ void layernorm_k(const float* __restrict__ x,
                            const float* __restrict__ g,
                            const float* __restrict__ b,
                            float* __restrict__ y, bf16_t* __restrict__ ybf)
{
  const int r = blockIdx.x;
  const float* xr = x + (size_t)r * 768;
  __shared__ float red[BDIM];
  float s = 0.f;
  for (int d = threadIdx.x; d < 768; d += BDIM) s += xr[d];
  red[threadIdx.x] = s; __syncthreads();
  for (int o = BDIM / 2; o > 0; o >>= 1) {
    if (threadIdx.x < o) red[threadIdx.x] += red[threadIdx.x + o];
    __syncthreads();
  }
  const float mean = red[0] / 768.0f;
  __syncthreads();
  float s2 = 0.f;
  for (int d = threadIdx.x; d < 768; d += BDIM) {
    const float v = xr[d] - mean; s2 += v * v;
  }
  red[threadIdx.x] = s2; __syncthreads();
  for (int o = BDIM / 2; o > 0; o >>= 1) {
    if (threadIdx.x < o) red[threadIdx.x] += red[threadIdx.x + o];
    __syncthreads();
  }
  const float rstd = rsqrtf(red[0] / 768.0f + 1e-5f);
  for (int d = threadIdx.x; d < 768; d += BDIM) {
    const float v = (xr[d] - mean) * rstd * g[d] + b[d];
    y[(size_t)r * 768 + d]   = v;
    ybf[(size_t)r * 768 + d] = f2bf(v);
  }
}

// Softmax over scores row (scale 1/8 + causal mask); stores P^T as bf16 so
// 'bhst,bshf->bthf' (sum over s) becomes an NN GEMM with K-contiguous A.
__global__ void softmax_T_k(const float* __restrict__ scores,
                            const float* __restrict__ mask,
                            bf16_t* __restrict__ pT, int S)
{
  const int r = blockIdx.x;          // r = (b*H + h)*S + s
  const int s = r % S;
  const int zz = r / S;
  const float* row  = scores + (size_t)r * S;
  const float* mrow = mask + (size_t)s * S;
  __shared__ float red[BDIM];

  float mx = -3.4e38f;
  for (int tt = threadIdx.x; tt < S; tt += BDIM)
    mx = fmaxf(mx, row[tt] * 0.125f + mrow[tt]);
  red[threadIdx.x] = mx; __syncthreads();
  for (int o = BDIM / 2; o > 0; o >>= 1) {
    if (threadIdx.x < o) red[threadIdx.x] = fmaxf(red[threadIdx.x], red[threadIdx.x + o]);
    __syncthreads();
  }
  mx = red[0]; __syncthreads();

  float sum = 0.f;
  for (int tt = threadIdx.x; tt < S; tt += BDIM)
    sum += __expf(row[tt] * 0.125f + mrow[tt] - mx);
  red[threadIdx.x] = sum; __syncthreads();
  for (int o = BDIM / 2; o > 0; o >>= 1) {
    if (threadIdx.x < o) red[threadIdx.x] += red[threadIdx.x + o];
    __syncthreads();
  }
  const float inv = 1.0f / red[0];
  bf16_t* pz = pT + (size_t)zz * S * S;
  for (int tt = threadIdx.x; tt < S; tt += BDIM) {
    const float v = __expf(row[tt] * 0.125f + mrow[tt] - mx) * inv;
    pz[(size_t)tt * S + s] = f2bf(v);     // transposed store
  }
}

// Per-(token,head) top-1 expert select + V projection (KA=1 => gate prob = 1).
// Stores V transposed: vselT[(b*12+h)*64+f][s]  (ld = 1024) so the attention
// GEMM's B operand is K-contiguous.
__global__ void vsel_k(const float* __restrict__ ln1,
                       const float* __restrict__ glog,
                       const float* __restrict__ Wv,
                       bf16_t* __restrict__ vselT, int* __restrict__ eidx,
                       float* __restrict__ cnt)
{
  const int t = blockIdx.x;
  const int bb = t >> 10, s = t & 1023;
  __shared__ float sx[768];
  __shared__ int se[12];
  for (int i = threadIdx.x; i < 768; i += BDIM) sx[i] = ln1[(size_t)t * 768 + i];
  if (threadIdx.x < 12) {
    const int h = threadIdx.x;
    const float* gl = glog + (size_t)t * 96 + h * 8;
    int best = 0; float bv = gl[0];
    for (int e = 1; e < 8; ++e) if (gl[e] > bv) { bv = gl[e]; best = e; }
    se[h] = best;
    eidx[t * 12 + h] = best;
    atomicAdd(&cnt[h * 8 + best], 1.0f);
  }
  __syncthreads();
  #pragma unroll
  for (int j = 0; j < 3; ++j) {
    const int idx = threadIdx.x + j * BDIM;     // 768 outputs
    const int h = idx >> 6, f = idx & 63;
    const int e = se[h];
    const float* W = Wv + ((size_t)(h * 8 + e) * 64) * 64;  // [h][e][d][f]
    float a = 0.f;
    for (int d = 0; d < 64; ++d) a += sx[h * 64 + d] * W[d * 64 + f];
    vselT[((size_t)(bb * 12 + h) * 64 + f) * 1024 + s] = f2bf(a);
  }
}

// O projection per (token,head) with same expert; x1 = x + attn_output.
__global__ void oproj_k(const float* __restrict__ x,
                        const float* __restrict__ attn,
                        const int* __restrict__ eidx,
                        const float* __restrict__ Wo,
                        float* __restrict__ x1)
{
  const int t = blockIdx.x;
  __shared__ float sa[768];
  for (int i = threadIdx.x; i < 768; i += BDIM) sa[i] = attn[(size_t)t * 768 + i];
  __syncthreads();
  #pragma unroll
  for (int j = 0; j < 3; ++j) {
    const int idx = threadIdx.x + j * BDIM;
    const int h = idx >> 6, g = idx & 63;
    const int e = eidx[t * 12 + h];
    const float* W = Wo + ((size_t)(h * 8 + e) * 64) * 64;  // [h][e][f][g]
    float a = 0.f;
    for (int f = 0; f < 64; ++f) a += sa[h * 64 + f] * W[f * 64 + g];
    x1[(size_t)t * 768 + idx] = x[(size_t)t * 768 + idx] + a;
  }
}

// FFN gate logits (E=8) + top-2 + softmax over top-2.
__global__ void gateff_top2_k(const float* __restrict__ ln2,
                              const float* __restrict__ gff,
                              int* __restrict__ tidx, float* __restrict__ tp)
{
  const int t = blockIdx.x;
  const int e = threadIdx.x >> 5, lane = threadIdx.x & 31;
  __shared__ float part[BDIM];
  __shared__ float lg[8];
  const float* xr = ln2 + (size_t)t * 768;
  float acc = 0.f;
  for (int d = lane; d < 768; d += 32) acc += xr[d] * gff[d * 8 + e];
  part[threadIdx.x] = acc; __syncthreads();
  if (threadIdx.x < 8) {
    float s = 0.f;
    for (int i = 0; i < 32; ++i) s += part[threadIdx.x * 32 + i];
    lg[threadIdx.x] = s;
  }
  __syncthreads();
  if (threadIdx.x == 0) {
    int i0 = 0; float v0 = lg[0];
    for (int q = 1; q < 8; ++q) if (lg[q] > v0) { v0 = lg[q]; i0 = q; }
    int i1 = -1; float v1 = -3.4e38f;
    for (int q = 0; q < 8; ++q) if (q != i0 && lg[q] > v1) { v1 = lg[q]; i1 = q; }
    const float p1 = __expf(v1 - v0);
    const float inv = 1.0f / (1.0f + p1);
    tidx[t * 2] = i0; tidx[t * 2 + 1] = i1;
    tp[t * 2] = inv;  tp[t * 2 + 1] = p1 * inv;
  }
}

// Sequential capacity cumsum (reference order), renormalize kept probs,
// build per-(token,expert) combine weights + counters.
__global__ void route_k(const int* __restrict__ tidx, const float* __restrict__ tp,
                        float* __restrict__ w, float* __restrict__ tctr,
                        float* __restrict__ ictr, int ntok, int cap)
{
  int cnt[8];
  for (int e = 0; e < 8; ++e) cnt[e] = 0;
  for (int t = 0; t < ntok; ++t) {
    int ee[2]; float pp[2];
    for (int s = 0; s < 2; ++s) {
      ee[s] = tidx[t * 2 + s];
      cnt[ee[s]]++;
      pp[s] = (cnt[ee[s]] <= cap) ? tp[t * 2 + s] : 0.0f;
    }
    const float inv = 1.0f / (pp[0] + pp[1] + 1e-9f);
    pp[0] *= inv; pp[1] *= inv;
    for (int s = 0; s < 2; ++s) {
      w[t * 8 + ee[s]] += pp[s];
      if (pp[s] > 0.f) tctr[ee[s]] += 1.0f;
      ictr[ee[s]] += pp[s];
    }
  }
}

// xe = ln2 * w[:, e]   (scaled BEFORE GEMM/ReLU, matching reference)
__global__ void scale_k(const float* __restrict__ x, const float* __restrict__ w,
                        bf16_t* __restrict__ y, int e, long n)
{
  long i = (long)blockIdx.x * blockDim.x + threadIdx.x;
  if (i < n) y[i] = f2bf(x[i] * w[(i / 768) * 8 + e]);
}

__global__ void final_add_k(const float* __restrict__ x1, const float* __restrict__ yf,
                            float* __restrict__ out, long n)
{
  long i = (long)blockIdx.x * blockDim.x + threadIdx.x;
  if (i < n) out[i] = x1[i] + yf[i];
}

// aux = aux1(attention load-balance) + aux2(ffn importance)
__global__ void aux_k(const float* __restrict__ cnt, const float* __restrict__ tctr,
                      const float* __restrict__ ictr, float* __restrict__ out)
{
  float ema[96], s = 0.f;
  for (int i = 0; i < 96; ++i) { ema[i] = cnt[i] / 2048.0f * 0.01f; s += ema[i]; }
  const float inv = 1.0f / (s + 1e-9f);
  float a1 = 0.f;
  for (int i = 0; i < 96; ++i) { const float p = ema[i] * inv; a1 += p * p; }
  a1 *= 96.0f;                               // * E * H
  float ts = 0.f, is = 0.f;
  for (int e = 0; e < 8; ++e) { ts += tctr[e]; is += ictr[e]; }
  float a2 = 0.f;
  for (int e = 0; e < 8; ++e) a2 += (tctr[e] / ts) * (ictr[e] / is);
  a2 *= 8.0f;                                // * E
  out[0] = a1 + a2;
}

// ---------------------------------------------------------------------------
// Host orchestration
// ---------------------------------------------------------------------------
static inline int eblocks(long n) { return (int)((n + BDIM - 1) / BDIM); }

extern "C" void kernel_launch(void* const* d_in, const int* in_sizes, int n_in,
                              void* d_out, int out_size, void* d_ws, size_t ws_size,
                              hipStream_t stream) {
  (void)in_sizes; (void)n_in; (void)out_size; (void)ws_size;

  const long BS  = 2048;            // B*S
  const long BSD = BS * 768;        // 1,572,864
  const int  S   = 1024;
  const long SS  = (long)S * S;

  const float* x      = (const float*)d_in[0];
  const float* mask   = (const float*)d_in[1];
  const float* W_q    = (const float*)d_in[2];
  const float* W_k    = (const float*)d_in[3];
  const float* W_v    = (const float*)d_in[4];
  const float* W_o    = (const float*)d_in[5];
  const float* W_rt   = (const float*)d_in[6];
  const float* g1     = (const float*)d_in[7];
  const float* b1     = (const float*)d_in[8];
  const float* g2     = (const float*)d_in[9];
  const float* b2     = (const float*)d_in[10];
  const float* gateff = (const float*)d_in[11];
  const float* W1     = (const float*)d_in[12];
  const float* W2     = (const float*)d_in[13];
  float* out = (float*)d_out;

  char* ws = (char*)d_ws;
  size_t off = 0;
  auto alloc = [&](size_t bytes) -> void* {
    void* p = ws + off;
    off = (off + bytes + 255) & ~(size_t)255;
    return p;
  };

  float*  ln1_f   = (float*)alloc(BSD * 4);
  bf16_t* ln1_bf  = (bf16_t*)alloc(BSD * 2);
  bf16_t* Wqt_bf  = (bf16_t*)alloc(768 * 768 * 2);        // [N,K]
  bf16_t* Wkt_bf  = (bf16_t*)alloc(768 * 768 * 2);
  bf16_t* Wrt_bf  = (bf16_t*)alloc(96 * 768 * 2);
  bf16_t* q_bf    = (bf16_t*)alloc(BSD * 2);
  bf16_t* k_bf    = (bf16_t*)alloc(BSD * 2);
  float*  glog    = (float*)alloc(BS * 96 * 4);
  float*  scores  = (float*)alloc(24 * SS * 4);
  bf16_t* pT_bf   = (bf16_t*)alloc(24 * SS * 2);
  bf16_t* vselT   = (bf16_t*)alloc(BSD * 2);              // [(b*12+h)*64+f][s]
  int*    eidx    = (int*)alloc(BS * 12 * 4);
  float*  cnt_he  = (float*)alloc(96 * 4);
  float*  attn_f  = (float*)alloc(BSD * 4);
  float*  x1      = (float*)alloc(BSD * 4);
  float*  ln2_f   = (float*)alloc(BSD * 4);
  bf16_t* ln2_bf  = (bf16_t*)alloc(BSD * 2);
  int*    tidx    = (int*)alloc(BS * 2 * 4);
  float*  tp      = (float*)alloc(BS * 2 * 4);
  float*  wcomb   = (float*)alloc(BS * 8 * 4);
  float*  tctr    = (float*)alloc(8 * 4);
  float*  ictr    = (float*)alloc(8 * 4);
  bf16_t* W1t_bf  = (bf16_t*)alloc((long)8 * 2048 * 768 * 2);  // per-e [2048,768]
  bf16_t* W2t_bf  = (bf16_t*)alloc((long)8 * 768 * 2048 * 2);  // per-e [768,2048]
  bf16_t* xe_bf   = (bf16_t*)alloc(BSD * 2);
  bf16_t* h_bf    = (bf16_t*)alloc(BS * 2048 * 2);
  float*  yf      = (float*)alloc(BSD * 4);

  // ---- zero accumulators ----
  zero_k<<<eblocks(BSD), BDIM, 0, stream>>>(yf, BSD);
  zero_k<<<1, BDIM, 0, stream>>>(cnt_he, 96);
  zero_k<<<1, BDIM, 0, stream>>>(tctr, 8);
  zero_k<<<1, BDIM, 0, stream>>>(ictr, 8);
  zero_k<<<eblocks(BS * 8), BDIM, 0, stream>>>(wcomb, BS * 8);

  // ---- weight conversions: fp32 [K,N] -> bf16 [N,K] ----
  cvt_t_k<<<eblocks(768 * 768), BDIM, 0, stream>>>(W_q, Wqt_bf, 768, 768, 1);
  cvt_t_k<<<eblocks(768 * 768), BDIM, 0, stream>>>(W_k, Wkt_bf, 768, 768, 1);
  cvt_t_k<<<eblocks(96 * 768), BDIM, 0, stream>>>(W_rt, Wrt_bf, 768, 96, 1);
  cvt_t_k<<<eblocks((long)8 * 2048 * 768), BDIM, 0, stream>>>(W1, W1t_bf, 768, 2048, 8);
  cvt_t_k<<<eblocks((long)8 * 768 * 2048), BDIM, 0, stream>>>(W2, W2t_bf, 2048, 768, 8);

  // ---- attention path ----
  layernorm_k<<<(int)BS, BDIM, 0, stream>>>(x, g1, b1, ln1_f, ln1_bf);

  // q / k projections: [2048 x 768] -> bf16
  {
    dim3 g((2048 / 128) * (768 / 64), 1, 1);
    gemm_lds<3><<<g, BDIM, 0, stream>>>(ln1_bf, Wqt_bf, q_bf,
        2048, 768, 768, 768, 768, 768, 1, 0, 0, 0, 0, 0, 0);
    gemm_lds<3><<<g, BDIM, 0, stream>>>(ln1_bf, Wkt_bf, k_bf,
        2048, 768, 768, 768, 768, 768, 1, 0, 0, 0, 0, 0, 0);
  }
  // router logits: [2048 x 96]  (N % 64 != 0 -> direct kernel)
  {
    dim3 g(((2048 / 32) * (96 / 32)) / (BDIM / 32), 1, 1);
    gemm_direct<0><<<g, BDIM, 0, stream>>>(ln1_bf, Wrt_bf, glog,
        2048, 96, 768, 768, 768, 96);
  }
  // scores[b,h] = q * k^T  (24 batched GEMMs, K=64): B operand k is [t, d]
  {
    dim3 g((1024 / 128) * (1024 / 64), 1, 24);
    gemm_lds<0><<<g, BDIM, 0, stream>>>(q_bf, k_bf, scores,
        1024, 1024, 64, 768, 768, 1024,
        12, (long)S * 768, 64, (long)S * 768, 64, 12 * SS, SS);
  }
  softmax_T_k<<<24 * S, BDIM, 0, stream>>>(scores, mask, pT_bf, S);

  vsel_k<<<(int)BS, BDIM, 0, stream>>>(ln1_f, glog, W_v, vselT, eidx, cnt_he);

  // attn_out[b,t,h,f] = sum_s P^T[t,s] V[s,f] : A = P^T, B = vselT [f][s]
  {
    dim3 g((1024 / 128) * (64 / 64), 1, 24);
    gemm_lds<0><<<g, BDIM, 0, stream>>>(pT_bf, vselT, attn_f,
        1024, 64, 1024, 1024, 1024, 768,
        12, 12 * SS, SS, (long)12 * 64 * 1024, (long)64 * 1024,
        (long)S * 768, 64);
  }
  oproj_k<<<(int)BS, BDIM, 0, stream>>>(x, attn_f, eidx, W_o, x1);

  // ---- MoE FFN path ----
  layernorm_k<<<(int)BS, BDIM, 0, stream>>>(x1, g2, b2, ln2_f, ln2_bf);
  gateff_top2_k<<<(int)BS, BDIM, 0, stream>>>(ln2_f, gateff, tidx, tp);
  route_k<<<1, 1, 0, stream>>>(tidx, tp, wcomb, tctr, ictr, (int)BS, 320);

  for (int e = 0; e < 8; ++e) {
    scale_k<<<eblocks(BSD), BDIM, 0, stream>>>(ln2_f, wcomb, xe_bf, e, BSD);
    // h = relu(xe @ W1[e]) -> bf16 ; W1t[e] is [2048, 768]
    {
      dim3 g((2048 / 128) * (2048 / 64), 1, 1);
      gemm_lds<2><<<g, BDIM, 0, stream>>>(xe_bf, W1t_bf + (size_t)e * 2048 * 768, h_bf,
          2048, 2048, 768, 768, 768, 2048, 1, 0, 0, 0, 0, 0, 0);
    }
    // yf += h @ W2[e] ; W2t[e] is [768, 2048]
    {
      dim3 g((2048 / 128) * (768 / 64), 1, 1);
      gemm_lds<1><<<g, BDIM, 0, stream>>>(h_bf, W2t_bf + (size_t)e * 768 * 2048, yf,
          2048, 768, 2048, 2048, 2048, 768, 1, 0, 0, 0, 0, 0, 0);
    }
  }

  // ---- outputs ----
  final_add_k<<<eblocks(BSD), BDIM, 0, stream>>>(x1, yf, out, BSD);
  aux_k<<<1, 1, 0, stream>>>(cnt_he, tctr, ictr, out + BSD);
}
